// VariationalDequantizer_88476326297717
// MI455X (gfx1250) — matrix operand order repaired and checked by reading, not compiled, not run on hardware
//
#include <hip/hip_runtime.h>
#include <hip/hip_bf16.h>
#include <math.h>

typedef __attribute__((ext_vector_type(16))) _Float16 v16h;
typedef __attribute__((ext_vector_type(8)))  float    v8f;
typedef __attribute__((ext_vector_type(4)))  unsigned int u32x4;
typedef __attribute__((ext_vector_type(8)))  unsigned int u32x8;

#define BSZ 16
#define NN  128
#define HID 64
#define LOG2PI 1.8378770664093453f

__device__ __forceinline__ float silu_f(float x) { return x / (1.0f + __expf(-x)); }
__device__ __forceinline__ float logsig_f(float x) {
    return (x < 0.0f) ? (x - log1pf(__expf(x))) : (-log1pf(__expf(-x)));
}
__device__ __forceinline__ v8f wmma_f16(v16h a, v16h b, v8f c) {
    return __builtin_amdgcn_wmma_f32_16x16x32_f16(false, a, false, b, (short)0, c, false, false);
}

// TDM: DMA a contiguous row of `nhalf` f16 elements from global into LDS.
__device__ __forceinline__ void tdm_load_row_f16(const _Float16* gsrc, _Float16* lds_dst,
                                                 unsigned nhalf) {
    unsigned long long ga = (unsigned long long)(uintptr_t)gsrc;
    unsigned lds_off = (unsigned)(uintptr_t)lds_dst;  // low 32 bits = LDS byte offset
    u32x4 g0;
    g0[0] = 1u;                                       // count = 1, user descriptor
    g0[1] = lds_off;                                  // lds_addr [63:32]
    g0[2] = (unsigned)(ga & 0xFFFFFFFFu);             // global_addr [95:64]
    g0[3] = (unsigned)((ga >> 32) & 0x01FFFFFFu) | 0x80000000u;  // addr[56:32] | type=2
    u32x8 g1;
    g1[0] = (1u << 16);                               // data_size = 1 (2 bytes)
    g1[1] = (nhalf & 0xFFFFu) << 16;                  // tensor_dim0[15:0] at bits 63:48
    g1[2] = (nhalf >> 16) | (1u << 16);               // tensor_dim0 hi | tensor_dim1 = 1
    g1[3] = (nhalf & 0xFFFFu) << 16;                  // tile_dim0 at bits 127:112
    g1[4] = 1u;                                       // tile_dim1 = 1
    g1[5] = nhalf;                                    // tensor_dim0_stride low 32
    g1[6] = 0u;
    g1[7] = 0u;
    asm volatile("tensor_load_to_lds %0, %1" :: "s"(g0), "s"(g1) : "memory");
}

// ---------------- zero log_qv slots ----------------
__global__ void zero_kernel(float* logqv) {
    if (threadIdx.x < BSZ) logqv[threadIdx.x] = 0.0f;
}

// ---------------- pre-transpose + f16-convert a 64x64 weight ----------------
__global__ void wprep_kernel(const float* __restrict__ W, _Float16* __restrict__ WT) {
    int t = blockIdx.x * 256 + threadIdx.x;  // 4096
    int n = t >> 6, k = t & 63;
    WT[n * 64 + k] = (_Float16)W[k * 64 + n];
}

// ---------------- radial + masked edge attr ----------------
__global__ void prep_kernel(const float* __restrict__ x, const float* __restrict__ nm,
                            const float* __restrict__ eattr, const float* __restrict__ emask,
                            float* __restrict__ radial, float* __restrict__ eab) {
    int idx = blockIdx.x * 256 + threadIdx.x;           // BS*N*N
    int b = idx >> 14, rem = idx & 16383;
    int i = rem >> 7, j = rem & 127;
    float nmi = nm[b * NN + i], nmj = nm[b * NN + j];
    float r2 = 0.0f;
#pragma unroll
    for (int d = 0; d < 3; d++) {
        float xi = x[(size_t)(b * NN + i) * 3 + d] * nmi;
        float xj = x[(size_t)(b * NN + j) * 3 + d] * nmj;
        float df = xi - xj;
        r2 += df * df;
    }
    radial[idx] = r2;
    eab[idx] = eattr[idx] * emask[idx];
}

// ---------------- embedding: h = (feat*nm) @ We + be ----------------
__global__ void emb_kernel(const float* __restrict__ int_f, const float* __restrict__ flt_f,
                           const float* __restrict__ nm, const float* __restrict__ We,
                           const float* __restrict__ be, float* __restrict__ h) {
    int idx = blockIdx.x * 256 + threadIdx.x;           // 2048*64
    int g = idx >> 6, c = idx & 63;
    float nmv = nm[g];
    float s = be[c];
#pragma unroll
    for (int k = 0; k < 8; k++) s += int_f[g * 8 + k] * nmv * We[k * 64 + c];
#pragma unroll
    for (int k = 0; k < 8; k++) s += flt_f[g * 8 + k] * nmv * We[(8 + k) * 64 + c];
    h[(size_t)g * 64 + c] = s;
}

// ---------------- Hi = h @ W1[0:64], Hj = h @ W1[64:128]  (WMMA) ----------------
__global__ void proj_kernel(const float* __restrict__ h, const float* __restrict__ W1,
                            float* __restrict__ Hi, float* __restrict__ Hj) {
    __shared__ __align__(32) _Float16 WaT[64 * 64];   // [n][k]
    __shared__ __align__(32) _Float16 WbT[64 * 64];
    int tid = threadIdx.x;
    for (int t = tid; t < 4096; t += 256) {
        int n = t & 63, k = t >> 6;
        WaT[n * 64 + k] = (_Float16)W1[k * 64 + n];
        WbT[n * 64 + k] = (_Float16)W1[(k + 64) * 64 + n];
    }
    __syncthreads();

    int wave = tid >> 5, lane = tid & 31;
    int rowA = blockIdx.x * 128 + wave * 16 + (lane & 15);
    int kb = (lane >> 4) * 8, kb2 = (lane >> 4) * 16;

    v8f acca[4] = {}, accb[4] = {};
#pragma unroll
    for (int ks = 0; ks < 2; ks++) {
        v16h a;
        const float* hr = &h[(size_t)rowA * 64 + ks * 32 + kb2];
#pragma unroll
        for (int e = 0; e < 8; e++) a[e] = (_Float16)hr[e];
#pragma unroll
        for (int e = 0; e < 8; e++) a[8 + e] = (_Float16)hr[8 + e];
#pragma unroll
        for (int nt = 0; nt < 4; nt++) {
            int n = nt * 16 + (lane & 15);
            v16h bfa = *(const v16h*)&WaT[n * 64 + ks * 32 + kb2];
            v16h bfb = *(const v16h*)&WbT[n * 64 + ks * 32 + kb2];
            acca[nt] = wmma_f16(a, bfa, acca[nt]);
            accb[nt] = wmma_f16(a, bfb, accb[nt]);
        }
    }
    int rowc = blockIdx.x * 128 + wave * 16 + (lane >> 4) * 8;
#pragma unroll
    for (int nt = 0; nt < 4; nt++) {
        int c = nt * 16 + (lane & 15);
#pragma unroll
        for (int r = 0; r < 8; r++) {
            Hi[(size_t)(rowc + r) * 64 + c] = acca[nt][r];
            Hj[(size_t)(rowc + r) * 64 + c] = accb[nt][r];
        }
    }
}

// ---------------- edge MLP: t=silu(Hi+Hj+rad*wr+ea*we+b1); m=silu(t@W2+b2)*em ----------------
__global__ void edge_kernel(const float* __restrict__ Hi, const float* __restrict__ Hj,
                            const float* __restrict__ radial, const float* __restrict__ eab,
                            const float* __restrict__ W1, const float* __restrict__ b1,
                            const _Float16* __restrict__ W2Tg, const float* __restrict__ b2,
                            const float* __restrict__ emask, float* __restrict__ Agg,
                            const float* __restrict__ eoW, const float* __restrict__ eob,
                            float* __restrict__ ebuf, int store_e) {
    __shared__ __align__(32) _Float16 Wt2[64 * 64];   // [n][k], staged by TDM
    __shared__ float HiB[64], wr[64], we[64], b2s[64], aggs[64], ems[128];
    __shared__ float weo0[64], weo1[64], beo[2];
    int tid = threadIdx.x;
    size_t ebase = (size_t)blockIdx.x * NN;            // (b*N+i)*N
    int b = blockIdx.x >> 7;

    // async DMA of the pre-converted f16 [n][k] weight tile into LDS
    tdm_load_row_f16(W2Tg, &Wt2[0], 4096u);

    if (tid < 64) {
        HiB[tid] = Hi[(size_t)blockIdx.x * 64 + tid] + b1[tid];
        wr[tid] = W1[128 * 64 + tid];
        we[tid] = W1[129 * 64 + tid];
        b2s[tid] = b2[tid];
        aggs[tid] = 0.0f;
        weo0[tid] = eoW[tid * 2 + 0];
        weo1[tid] = eoW[tid * 2 + 1];
    }
    if (tid < 128) ems[tid] = emask[ebase + tid];
    if (tid < 2) beo[tid] = eob[tid];

    int wave = tid >> 5, lane = tid & 31;
    int mbase = wave * 16;
    int jrow = mbase + (lane & 15);
    int kb = (lane >> 4) * 8, kb2 = (lane >> 4) * 16, hl = lane >> 4;
    const float* HjR = &Hj[(size_t)(b * NN + jrow) * 64];
    __builtin_prefetch(HjR, 0, 0);
    float rad = radial[ebase + jrow];
    float eav = eab[ebase + jrow];

    __builtin_amdgcn_s_wait_tensorcnt(0);
    __syncthreads();

    v8f acc[4] = {};
#pragma unroll
    for (int ks = 0; ks < 2; ks++) {
        v16h a;
        int c0 = ks * 32 + kb2;
#pragma unroll
        for (int e = 0; e < 8; e++) {
            int c = c0 + e;
            float v = HiB[c] + HjR[c] + rad * wr[c] + eav * we[c];
            a[e] = (_Float16)silu_f(v);
        }
#pragma unroll
        for (int e = 0; e < 8; e++) {
            int c = c0 + 8 + e;
            float v = HiB[c] + HjR[c] + rad * wr[c] + eav * we[c];
            a[8 + e] = (_Float16)silu_f(v);
        }
#pragma unroll
        for (int nt = 0; nt < 4; nt++) {
            int n = nt * 16 + (lane & 15);
            v16h bf = *(const v16h*)&Wt2[n * 64 + ks * 32 + kb2];
            acc[nt] = wmma_f16(a, bf, acc[nt]);
        }
    }

    // m = silu(acc + b2) * em ; scatter-sum into aggs
    float mv[4][8];
#pragma unroll
    for (int nt = 0; nt < 4; nt++) {
        int c = nt * 16 + (lane & 15);
        float bb = b2s[c];
        float s = 0.0f;
#pragma unroll
        for (int r = 0; r < 8; r++) {
            int j = mbase + r + 8 * hl;
            float v = silu_f(acc[nt][r] + bb) * ems[j];
            mv[nt][r] = v;
            s += v;
        }
        atomicAdd(&aggs[c], s);
    }

    if (store_e) {
#pragma unroll
        for (int r = 0; r < 8; r++) {
            float p0 = 0.0f, p1 = 0.0f;
#pragma unroll
            for (int nt = 0; nt < 4; nt++) {
                int cc = nt * 16 + (lane & 15);
                p0 += mv[nt][r] * weo0[cc];
                p1 += mv[nt][r] * weo1[cc];
            }
#pragma unroll
            for (int m = 1; m < 16; m <<= 1) {
                p0 += __shfl_xor(p0, m, 32);
                p1 += __shfl_xor(p1, m, 32);
            }
            if ((lane & 15) == 0) {
                int j = mbase + r + 8 * hl;
                ebuf[(ebase + j) * 2 + 0] = (p0 + beo[0]) * ems[j];
                ebuf[(ebase + j) * 2 + 1] = (p1 + beo[1]) * ems[j];
            }
        }
    }
    __syncthreads();
    if (tid < 64) Agg[(size_t)blockIdx.x * 64 + tid] = aggs[tid];
}

// ---------------- node MLP: h += [h|agg]@Wn1 -> silu -> @Wn2 + bn2 ----------------
__global__ void node_kernel(float* __restrict__ h, const float* __restrict__ Agg,
                            const float* __restrict__ Wn1, const float* __restrict__ bn1,
                            const float* __restrict__ Wn2, const float* __restrict__ bn2) {
    __shared__ __align__(32) _Float16 W1T[64 * 128];  // [n][k], k = 0..127
    __shared__ __align__(32) _Float16 W2T[64 * 64];
    __shared__ __align__(32) _Float16 T1[8 * 16 * 64];
    __shared__ float b1s[64], b2s[64];
    int tid = threadIdx.x;
    for (int t = tid; t < 8192; t += 256) {
        int n = t >> 7, k = t & 127;
        W1T[n * 128 + k] = (_Float16)Wn1[k * 64 + n];
    }
    for (int t = tid; t < 4096; t += 256) {
        int n = t >> 6, k = t & 63;
        W2T[n * 64 + k] = (_Float16)Wn2[k * 64 + n];
    }
    if (tid < 64) { b1s[tid] = bn1[tid]; b2s[tid] = bn2[tid]; }
    __syncthreads();

    int wave = tid >> 5, lane = tid & 31;
    int rowA = blockIdx.x * 128 + wave * 16 + (lane & 15);
    int kb = (lane >> 4) * 8, kb2 = (lane >> 4) * 16, hl = lane >> 4;

    v8f acc[4] = {};
#pragma unroll
    for (int ks = 0; ks < 4; ks++) {
        v16h a;
#pragma unroll
        for (int e = 0; e < 16; e++) {
            int c = ks * 32 + kb2 + e;
            float v = (c < 64) ? h[(size_t)rowA * 64 + c] : Agg[(size_t)rowA * 64 + (c - 64)];
            a[e] = (_Float16)v;
        }
#pragma unroll
        for (int nt = 0; nt < 4; nt++) {
            int n = nt * 16 + (lane & 15);
            v16h bf = *(const v16h*)&W1T[n * 128 + ks * 32 + kb2];
            acc[nt] = wmma_f16(a, bf, acc[nt]);
        }
    }

    _Float16* tw = &T1[wave * 16 * 64];
#pragma unroll
    for (int nt = 0; nt < 4; nt++) {
        int c = nt * 16 + (lane & 15);
        float bb = b1s[c];
#pragma unroll
        for (int r = 0; r < 8; r++)
            tw[(r + 8 * hl) * 64 + c] = (_Float16)silu_f(acc[nt][r] + bb);
    }
    __syncthreads();

    v8f acc2[4] = {};
#pragma unroll
    for (int ks = 0; ks < 2; ks++) {
        v16h a;
        const _Float16* ar = &tw[(lane & 15) * 64 + ks * 32 + kb2];
#pragma unroll
        for (int e = 0; e < 8; e++) a[e] = ar[e];
#pragma unroll
        for (int e = 0; e < 8; e++) a[8 + e] = ar[8 + e];
#pragma unroll
        for (int nt = 0; nt < 4; nt++) {
            int n = nt * 16 + (lane & 15);
            v16h bf = *(const v16h*)&W2T[n * 64 + ks * 32 + kb2];
            acc2[nt] = wmma_f16(a, bf, acc2[nt]);
        }
    }
    int rowc = blockIdx.x * 128 + wave * 16 + hl * 8;
#pragma unroll
    for (int nt = 0; nt < 4; nt++) {
        int c = nt * 16 + (lane & 15);
#pragma unroll
        for (int r = 0; r < 8; r++) {
            size_t off = (size_t)(rowc + r) * 64 + c;
            h[off] = h[off] + acc2[nt][r] + b2s[c];
        }
    }
}

// ---------------- node head + affine/sigmoid flow ----------------
__global__ void hflow_kernel(const float* __restrict__ h, const float* __restrict__ Wout,
                             const float* __restrict__ bout, const float* __restrict__ nm,
                             const float* __restrict__ int_feat, const float* __restrict__ h_eps,
                             float* __restrict__ v_int, float* __restrict__ logqv) {
    int g = blockIdx.x * 256 + threadIdx.x;  // 2048 nodes
    int b = g >> 7;
    float nmv = nm[g];
    float ho[16];
#pragma unroll
    for (int d = 0; d < 16; d++) ho[d] = bout[d];
    for (int c = 0; c < 64; c++) {
        float hv = h[(size_t)g * 64 + c];
#pragma unroll
        for (int d = 0; d < 16; d++) ho[d] += hv * Wout[c * 16 + d];
    }
    float acc = 0.0f;
#pragma unroll
    for (int d = 0; d < 8; d++) {
        float mu = ho[d] * nmv, ls = ho[d + 8] * nmv;
        float heps = h_eps[g * 8 + d] * nmv;
        float hu = mu + heps * __expf(ls);
        float ui = 1.0f / (1.0f + __expf(-hu));
        v_int[g * 8 + d] = (int_feat[g * 8 + d] + ui) * nmv;
        acc += nmv * (-0.5f * heps * heps - 0.5f * LOG2PI) - ls
             - nmv * (logsig_f(hu) + logsig_f(-hu));
    }
    for (int o = 16; o > 0; o >>= 1) acc += __shfl_down(acc, o, 32);
    if ((threadIdx.x & 31) == 0) atomicAdd(&logqv[b], acc);
}

// ---------------- edge flow ----------------
__global__ void eflow_kernel(const float* __restrict__ ebuf, const float* __restrict__ emask,
                             const float* __restrict__ e_eps, const float* __restrict__ eattr,
                             float* __restrict__ v_edge, float* __restrict__ logqv) {
    int idx = blockIdx.x * 256 + threadIdx.x;  // 262144
    int b = idx >> 14;
    float em = emask[idx];
    float mu = ebuf[(size_t)idx * 2 + 0], ls = ebuf[(size_t)idx * 2 + 1];
    float eeps = e_eps[idx] * em;
    float eu = mu + eeps * __expf(ls);
    float ue = 1.0f / (1.0f + __expf(-eu));
    v_edge[idx] = (eattr[idx] + ue) * em;
    float acc = em * (-0.5f * eeps * eeps - 0.5f * LOG2PI) - ls
              - em * (logsig_f(eu) + logsig_f(-eu));
    for (int o = 16; o > 0; o >>= 1) acc += __shfl_down(acc, o, 32);
    if ((threadIdx.x & 31) == 0) atomicAdd(&logqv[b], acc);
}

extern "C" void kernel_launch(void* const* d_in, const int* in_sizes, int n_in,
                              void* d_out, int out_size, void* d_ws, size_t ws_size,
                              hipStream_t stream) {
    const float* x        = (const float*)d_in[0];
    const float* int_feat = (const float*)d_in[1];
    const float* flt_feat = (const float*)d_in[2];
    const float* eattr    = (const float*)d_in[3];
    const float* nmsk     = (const float*)d_in[4];
    const float* emsk     = (const float*)d_in[5];
    const float* h_eps    = (const float*)d_in[6];
    const float* e_eps    = (const float*)d_in[7];
    // params flattened alphabetically: emb, eout, layers[0..3]{e1,e2,n1,n2}, out
    const float* embW  = (const float*)d_in[8];
    const float* embB  = (const float*)d_in[9];
    const float* eoutW = (const float*)d_in[10];
    const float* eoutB = (const float*)d_in[11];
    const float* outW  = (const float*)d_in[44];
    const float* outB  = (const float*)d_in[45];

    float* ws     = (float*)d_ws;
    float* h      = ws;                 // 2048*64
    float* Hi     = ws + 131072;        // 2048*64
    float* Hj     = ws + 262144;        // 2048*64
    float* Agg    = ws + 393216;        // 2048*64
    float* radial = ws + 524288;        // 262144
    float* eab    = ws + 786432;        // 262144
    float* ebuf   = ws + 1048576;       // 262144*2
    _Float16* w2t = (_Float16*)(ws + 1572864);  // 4 layers * 4096 halfs

    float* v_int  = (float*)d_out;                  // 16384
    float* v_edge = v_int + 16384;                  // 262144
    float* logqv  = v_int + 16384 + 262144;         // 16

    zero_kernel<<<1, 32, 0, stream>>>(logqv);
    prep_kernel<<<1024, 256, 0, stream>>>(x, nmsk, eattr, emsk, radial, eab);
    emb_kernel<<<512, 256, 0, stream>>>(int_feat, flt_feat, nmsk, embW, embB, h);
    for (int L = 0; L < 4; L++)
        wprep_kernel<<<16, 256, 0, stream>>>((const float*)d_in[14 + 8 * L], w2t + L * 4096);

    for (int L = 0; L < 4; L++) {
        const float* e1W = (const float*)d_in[12 + 8 * L];
        const float* e1B = (const float*)d_in[13 + 8 * L];
        const float* e2B = (const float*)d_in[15 + 8 * L];
        const float* n1W = (const float*)d_in[16 + 8 * L];
        const float* n1B = (const float*)d_in[17 + 8 * L];
        const float* n2W = (const float*)d_in[18 + 8 * L];
        const float* n2B = (const float*)d_in[19 + 8 * L];

        proj_kernel<<<16, 256, 0, stream>>>(h, e1W, Hi, Hj);
        edge_kernel<<<2048, 256, 0, stream>>>(Hi, Hj, radial, eab, e1W, e1B,
                                              w2t + L * 4096, e2B,
                                              emsk, Agg, eoutW, eoutB, ebuf, (L == 3) ? 1 : 0);
        node_kernel<<<16, 256, 0, stream>>>(h, Agg, n1W, n1B, n2W, n2B);
    }

    hflow_kernel<<<8, 256, 0, stream>>>(h, outW, outB, nmsk, int_feat, h_eps, v_int, logqv);
    eflow_kernel<<<1024, 256, 0, stream>>>(ebuf, emsk, e_eps, eattr, v_edge, logqv);
}
